// InvertedDispatchExpertBank_40664750359236
// MI455X (gfx1250) — compile-verified, matrix operand order checked by reading, not run on hardware
//
#include <hip/hip_runtime.h>
#include <hip/hip_bf16.h>

#define N_EXPERTS 64
#define K_ACTIVE  2
#define D_MODEL   512
#define D_FF      2048
#define CAPACITY  256
#define N_TOKENS  4096
#define S_ASSIGN  (N_TOKENS * K_ACTIVE)
#define KC        64      // K-chunk per LDS stage
#define NBLK      256     // N columns per workgroup (8 waves x 2 tiles of 16)
#define LDP       72      // padded LDS row stride (72*2B = 144B, 16B-aligned, bank-conflict-free)

typedef __attribute__((ext_vector_type(16))) __bf16 v16bf;
typedef __attribute__((ext_vector_type(8)))  __bf16 v8bf;
typedef __attribute__((ext_vector_type(4)))  __bf16 v4bf;
typedef __attribute__((ext_vector_type(8)))  float  v8f;

// ---------------- routing / dispatch ----------------

__global__ void moe_zero_kernel(float* out, int out_n, int* counts) {
  int i = blockIdx.x * blockDim.x + threadIdx.x;
  if (i < out_n) out[i] = 0.0f;
  if (i < N_EXPERTS) counts[i] = 0;
}

__global__ void moe_route_kernel(const int* __restrict__ sel,
                                 int* __restrict__ counts,
                                 int* __restrict__ tokmap) {
  int s = blockIdx.x * blockDim.x + threadIdx.x;
  if (s >= S_ASSIGN) return;
  int e = sel[s];
  int r = atomicAdd(&counts[e], 1);
  if (r < CAPACITY) tokmap[e * CAPACITY + r] = s;   // s encodes (token, slot)
}

// one block (128 threads) per (expert, capacity-row); float4 -> v4bf
__global__ void moe_gather_kernel(const float* __restrict__ x,
                                  const int* __restrict__ counts,
                                  const int* __restrict__ tokmap,
                                  __bf16* __restrict__ Xg) {
  int b = blockIdx.x;                 // b = e*CAPACITY + r
  int e = b >> 8;
  int r = b & (CAPACITY - 1);
  int cnt = min(counts[e], CAPACITY);
  __bf16* dst = Xg + (size_t)b * D_MODEL;
  int c = threadIdx.x * 4;            // 128 threads * 4 = 512
  if (r < cnt) {
    const float* src = x + (size_t)(tokmap[b] >> 1) * D_MODEL;
    float4 f = *(const float4*)(src + c);
    v4bf p;
    p[0] = (__bf16)f.x; p[1] = (__bf16)f.y; p[2] = (__bf16)f.z; p[3] = (__bf16)f.w;
    *(v4bf*)(dst + c) = p;
  } else {
    v4bf z;
    z[0] = z[1] = z[2] = z[3] = (__bf16)0.0f;
    *(v4bf*)(dst + c) = z;
  }
}

__global__ void moe_loads_kernel(const int* __restrict__ counts, float* __restrict__ out_tail) {
  int e = threadIdx.x;
  if (e < N_EXPERTS) out_tail[e] = (float)counts[e] * (1.0f / (float)N_TOKENS);
}

// ---------------- WMMA fragment load (bf16 16x32 layout) ----------------
// lane L: row = L%16; K-half = L/16.  lo = K[kb..kb+7], hi = K[kb+16..kb+23],
// matching the 16-bit A/B VGPR layout (lanes 0-15: K 0-7 & 16-23; lanes 16-31: K 8-15 & 24-31).
__device__ __forceinline__ v16bf load_frag(const __bf16* rowbase, int kb) {
  v8bf lo = *(const v8bf*)(rowbase + kb);        // ds_load_b128
  v8bf hi = *(const v8bf*)(rowbase + kb + 16);   // ds_load_b128
  return __builtin_shufflevector(lo, hi, 0,1,2,3,4,5,6,7,8,9,10,11,12,13,14,15);
}

// ---------------- GEMM1: H = gelu(Xg @ W1[e]) ----------------
// grid = (8 n-blocks of 256, 4 m-blocks of 64, 64 experts), 256 threads = 8 waves.
__global__ __launch_bounds__(256) void moe_gemm1_kernel(const float* __restrict__ W1,
                                                        const __bf16* __restrict__ Xg,
                                                        const int* __restrict__ counts,
                                                        __bf16* __restrict__ H) {
  const int e = blockIdx.z, mb = blockIdx.y, nb = blockIdx.x;
  const int cnt = min(counts[e], CAPACITY);
  if (mb * 64 >= cnt) return;                    // skip empty M-blocks (uniform)

  __shared__ __align__(16) __bf16 Wt[NBLK][LDP]; // W chunk transposed: [n][k]
  __shared__ __align__(16) __bf16 Xs[64][LDP];   // X chunk: [m][k]

  const int tid   = threadIdx.x;
  const int lane  = tid & 31;
  const int w     = tid >> 5;        // wave id 0..7 -> N cols w*16 and 128+w*16
  const int r16   = lane & 15;
  const int khalf = lane >> 4;

  const float*  W1e = W1 + (size_t)e * D_MODEL * D_FF + (size_t)nb * NBLK;
  const __bf16* Xge = Xg + ((size_t)e * CAPACITY + (size_t)mb * 64) * D_MODEL;

  v8f acc[4][2] = {};

  for (int kc = 0; kc < D_MODEL; kc += KC) {
    if (kc + KC < D_MODEL)   // prefetch next weight chunk (global_prefetch_b8)
      __builtin_prefetch(W1e + (size_t)(kc + KC + (tid & 63)) * D_FF + (tid >> 6) * 64, 0, 1);
    // stage W chunk (64k x 256n), fp32 -> bf16, transposed; 4 k's packed per ds_store_b64
#pragma unroll 4
    for (int it = 0; it < 16; ++it) {
      int idx = it * 256 + tid;
      int n = idx & 255, k = (idx >> 8) * 4;
      const float* g = W1e + (size_t)(kc + k) * D_FF + n;
      v4bf p;
      p[0] = (__bf16)g[0];
      p[1] = (__bf16)g[(size_t)D_FF];
      p[2] = (__bf16)g[(size_t)2 * D_FF];
      p[3] = (__bf16)g[(size_t)3 * D_FF];
      *(v4bf*)&Wt[n][k] = p;
    }
    // stage X chunk (64m x 64k), bf16 16B copies (ds_store_b128)
#pragma unroll
    for (int it = 0; it < 2; ++it) {
      int idx = it * 256 + tid;
      int kv = (idx & 7) * 8, m = idx >> 3;
      *(v8bf*)&Xs[m][kv] = *(const v8bf*)(Xge + (size_t)m * D_MODEL + kc + kv);
    }
    __syncthreads();
#pragma unroll
    for (int ks = 0; ks < 2; ++ks) {
      const int kb = ks * 32 + khalf * 8;
      v16bf b0 = load_frag(&Wt[w * 16 + r16][0], kb);
      v16bf b1 = load_frag(&Wt[128 + w * 16 + r16][0], kb);
#pragma unroll
      for (int mt = 0; mt < 4; ++mt) {
        v16bf a = load_frag(&Xs[mt * 16 + r16][0], kb);
        acc[mt][0] = __builtin_amdgcn_wmma_f32_16x16x32_bf16(
            false, a, false, b0, (short)0, acc[mt][0], false, false);
        acc[mt][1] = __builtin_amdgcn_wmma_f32_16x16x32_bf16(
            false, a, false, b1, (short)0, acc[mt][1], false, false);
      }
    }
    __syncthreads();
  }

  // exact-GELU epilogue, store bf16 H.  C layout: VGPR v -> M = v + 8*(lane>=16), N = lane%16.
  __bf16* He = H + (size_t)e * CAPACITY * D_FF;
  const int mrow0 = mb * 64 + 8 * khalf;
  const int ncol0 = nb * NBLK + w * 16 + r16;
#pragma unroll
  for (int mt = 0; mt < 4; ++mt) {
#pragma unroll
    for (int j = 0; j < 2; ++j) {
#pragma unroll
      for (int v = 0; v < 8; ++v) {
        float xv = acc[mt][j][v];
        float g  = 0.5f * xv * (1.0f + erff(xv * 0.70710678118654752f));
        He[(size_t)(mrow0 + mt * 16 + v) * D_FF + ncol0 + j * 128] = (__bf16)g;
      }
    }
  }
}

// ---------------- GEMM2: out[tok,slot] = H @ W2[e] ----------------
// grid = (2 n-blocks of 256, 4 m-blocks of 64, 64 experts), 256 threads.
__global__ __launch_bounds__(256) void moe_gemm2_kernel(const float* __restrict__ W2,
                                                        const __bf16* __restrict__ H,
                                                        const int* __restrict__ counts,
                                                        const int* __restrict__ tokmap,
                                                        float* __restrict__ out) {
  const int e = blockIdx.z, mb = blockIdx.y, nb = blockIdx.x;
  const int cnt = min(counts[e], CAPACITY);
  if (mb * 64 >= cnt) return;

  __shared__ __align__(16) __bf16 Wt[NBLK][LDP];
  __shared__ __align__(16) __bf16 Xs[64][LDP];

  const int tid   = threadIdx.x;
  const int lane  = tid & 31;
  const int w     = tid >> 5;
  const int r16   = lane & 15;
  const int khalf = lane >> 4;

  const float*  W2e = W2 + (size_t)e * D_FF * D_MODEL + (size_t)nb * NBLK;
  const __bf16* He  = H + ((size_t)e * CAPACITY + (size_t)mb * 64) * D_FF;

  v8f acc[4][2] = {};

  for (int kc = 0; kc < D_FF; kc += KC) {
    if (kc + KC < D_FF)
      __builtin_prefetch(W2e + (size_t)(kc + KC + (tid & 63)) * D_MODEL + (tid >> 6) * 64, 0, 1);
#pragma unroll 4
    for (int it = 0; it < 16; ++it) {
      int idx = it * 256 + tid;
      int n = idx & 255, k = (idx >> 8) * 4;
      const float* g = W2e + (size_t)(kc + k) * D_MODEL + n;
      v4bf p;
      p[0] = (__bf16)g[0];
      p[1] = (__bf16)g[(size_t)D_MODEL];
      p[2] = (__bf16)g[(size_t)2 * D_MODEL];
      p[3] = (__bf16)g[(size_t)3 * D_MODEL];
      *(v4bf*)&Wt[n][k] = p;
    }
#pragma unroll
    for (int it = 0; it < 2; ++it) {
      int idx = it * 256 + tid;
      int kv = (idx & 7) * 8, m = idx >> 3;
      *(v8bf*)&Xs[m][kv] = *(const v8bf*)(He + (size_t)m * D_FF + kc + kv);
    }
    __syncthreads();
#pragma unroll
    for (int ks = 0; ks < 2; ++ks) {
      const int kb = ks * 32 + khalf * 8;
      v16bf b0 = load_frag(&Wt[w * 16 + r16][0], kb);
      v16bf b1 = load_frag(&Wt[128 + w * 16 + r16][0], kb);
#pragma unroll
      for (int mt = 0; mt < 4; ++mt) {
        v16bf a = load_frag(&Xs[mt * 16 + r16][0], kb);
        acc[mt][0] = __builtin_amdgcn_wmma_f32_16x16x32_bf16(
            false, a, false, b0, (short)0, acc[mt][0], false, false);
        acc[mt][1] = __builtin_amdgcn_wmma_f32_16x16x32_bf16(
            false, a, false, b1, (short)0, acc[mt][1], false, false);
      }
    }
    __syncthreads();
  }

  // scatter-combine: row r -> assignment s -> out[s*512 + col]
  const int ncol0 = nb * NBLK + w * 16 + r16;
#pragma unroll
  for (int mt = 0; mt < 4; ++mt) {
#pragma unroll
    for (int v = 0; v < 8; ++v) {
      int r = mb * 64 + mt * 16 + v + 8 * khalf;
      if (r < cnt) {
        int s = tokmap[e * CAPACITY + r];
        out[(size_t)s * D_MODEL + ncol0]       = acc[mt][0][v];
        out[(size_t)s * D_MODEL + ncol0 + 128] = acc[mt][1][v];
      }
    }
  }
}

// ---------------- launch ----------------

extern "C" void kernel_launch(void* const* d_in, const int* in_sizes, int n_in,
                              void* d_out, int out_size, void* d_ws, size_t ws_size,
                              hipStream_t stream) {
  const float* x   = (const float*)d_in[0];   // hidden_states [4096,512]
  const int*   sel = (const int*)d_in[1];     // selected_experts [4096,2]
  // d_in[2] expert_masks: unused
  const float* W1  = (const float*)d_in[3];   // [64,512,2048]
  const float* W2  = (const float*)d_in[4];   // [64,2048,512]
  float* out = (float*)d_out;                 // [4096,2,512] + [64]

  char* ws = (char*)d_ws;
  int*    counts = (int*)ws;                                 // 256 B
  int*    tokmap = (int*)(ws + 256);                         // 64 KB
  __bf16* Xg     = (__bf16*)(ws + 256 + 64 * 1024);          // 16 MB  (bf16 [64,256,512])
  __bf16* H      = (__bf16*)(ws + 256 + 64 * 1024 + (size_t)16 * 1024 * 1024); // 64 MB

  moe_zero_kernel<<<(out_size + 255) / 256, 256, 0, stream>>>(out, out_size, counts);
  moe_route_kernel<<<S_ASSIGN / 256, 256, 0, stream>>>(sel, counts, tokmap);
  moe_gather_kernel<<<N_EXPERTS * CAPACITY, 128, 0, stream>>>(x, counts, tokmap, Xg);
  moe_loads_kernel<<<1, 64, 0, stream>>>(counts, out + (size_t)N_TOKENS * K_ACTIVE * D_MODEL);
  moe_gemm1_kernel<<<dim3(D_FF / NBLK, CAPACITY / 64, N_EXPERTS), 256, 0, stream>>>(W1, Xg, counts, H);
  moe_gemm2_kernel<<<dim3(D_MODEL / NBLK, CAPACITY / 64, N_EXPERTS), 256, 0, stream>>>(W2, H, counts, tokmap, out);
}
